// CorrTorch_43138651521622
// MI455X (gfx1250) — matrix operand order repaired and checked by reading, compiled, not run on hardware
//
#include <hip/hip_runtime.h>

// CDNA5 (gfx1250) FlowNet correlation via v_wmma_f32_16x16x32_bf16.
// out[b, dy*9+dx, y, x] = (1/128) * sum_c in1[b,c,y,x] * in2pad[b,c,y+dy,x+dx]
//
// Per workgroup = one (b, y) output row. 10 waves, one 16-pixel tile each.
// Banded matmul: two 16x16x32 WMMA N-tiles per wave cover the +-4 x-band;
// results are transposed through LDS so global writes are fully coalesced.

typedef __attribute__((ext_vector_type(16))) __bf16 v16bf;
typedef __attribute__((ext_vector_type(8)))  __bf16 bf16x8;   // 16 bytes
typedef __attribute__((ext_vector_type(8)))  float  v8f;
typedef __attribute__((ext_vector_type(4)))  float  f32x4;    // 16 bytes

#define C_CH     128
#define H_DIM    128
#define W_DIM    160
#define MD       4
#define ND       9
#define NPOS     172      // in2 row positions: x = pos - 4, pos in [0,172)
#define LSTR     136      // bf16 elements per position row (272B, 16B-aligned)
#define CT_COLS  28       // per-wave C-tile columns (x0-4 .. x0+23)
#define CT_STR   20       // floats per C-tile column (bank-conflict-free, 16B-able)
#define CT_WAVE  (CT_COLS * CT_STR)   // 560 floats per wave
#define NTHREADS 320      // 10 waves -> 10 x 16-pixel tiles = W_DIM

__global__ __launch_bounds__(NTHREADS)
void corr_wmma_kernel(const float* __restrict__ in1,
                      const float* __restrict__ in2,
                      float* __restrict__ out)
{
    __shared__ __align__(16) __bf16 Abuf[W_DIM * LSTR];       // in1 row, [pix][ch]
    __shared__ __align__(16) __bf16 Bbuf[NPOS * LSTR];        // in2 row + halo, [pos][ch]
    __shared__ __align__(16) float  Ctile[10 * CT_WAVE];      // band transpose buffer

    const int y    = blockIdx.x;          // 0..127
    const int b    = blockIdx.y;          // 0..15
    const int tid  = threadIdx.x;
    const int lane = tid & 31;
    const int wave = tid >> 5;            // 0..9
    const int x0   = wave * 16;
    const int HW   = H_DIM * W_DIM;

    // ---------------- stage in1 row y: fp32 NCHW -> bf16 [pix][ch] ----------------
    // Thread handles a 4-pixel chunk of a channel pair: 2x global_load_b128.
    {
        const float* g1 = in1 + (size_t)b * C_CH * HW + (size_t)y * W_DIM;
        for (int i = tid; i < 40 * (C_CH / 2); i += NTHREADS) {
            int chunk = i % 40;           // 4-pixel chunk, coalesced along x
            int cp    = i / 40;           // channel pair
            int x     = chunk * 4;
            f32x4 a0 = *(const f32x4*)(g1 + (2 * cp)     * HW + x);
            f32x4 a1 = *(const f32x4*)(g1 + (2 * cp + 1) * HW + x);
#pragma unroll
            for (int e = 0; e < 4; ++e) {
                union { __bf16 h[2]; unsigned u; } pk;
                pk.h[0] = (__bf16)a0[e];
                pk.h[1] = (__bf16)a1[e];
                *(unsigned*)&Abuf[(x + e) * LSTR + 2 * cp] = pk.u;   // 4B-aligned
            }
        }
    }
    __syncthreads();

    // ---------------- load A fragments once, keep across dy loop ----------------
    // A 16x32 as loaded here (same k order as B): lane m (and m+16) holds row m; one contiguous
    // 16-element K run at koff = (lane>=16)*16.
    v16bf afrag[4];
    {
        const __bf16* arow  = Abuf + (size_t)(x0 + (lane & 15)) * LSTR;
        const int     koffa = ((lane >> 4) & 1) * 16;
#pragma unroll
        for (int kc = 0; kc < 4; ++kc) {
            union { v16bf v; bf16x8 h[2]; } u;
            u.h[0] = *(const bf16x8*)(arow + kc * 32 + koffa);
            u.h[1] = *(const bf16x8*)(arow + kc * 32 + koffa + 8);
            afrag[kc] = u.v;
        }
    }

    const int   n     = lane & 15;
    const int   madd  = (lane >> 4) * 8;
    const float scale = 1.0f / (float)C_CH;

    // Fixed output mapping per thread (loop-invariant addresses).
    const int xo = tid % W_DIM;           // 0..159, coalesced along x
    const int sp = tid / W_DIM;           // plane parity: 0 or 1
    const float* ctp = Ctile + (xo >> 4) * CT_WAVE + (xo & 15) * CT_STR + (xo & 15);

    for (int dy = 0; dy < ND; ++dy) {
        __syncthreads();   // WAR guard on Bbuf/Ctile vs previous iteration

        // ---------------- stage in2 row (y+dy-4), x in [-4,168), zero-padded ----
        // 43 chunks of 4 positions; chunks 0, 41, 42 are fully out-of-range pad.
        const int  y2    = y + dy - MD;
        const bool rowok = (y2 >= 0) && (y2 < H_DIM);
        const float* g2  = in2 + (size_t)b * C_CH * HW + (size_t)y2 * W_DIM;
        for (int i = tid; i < 43 * (C_CH / 2); i += NTHREADS) {
            int chunk = i % 43;
            int cp    = i / 43;
            int x     = chunk * 4 - MD;   // -4, 0, 4, ..., 164 (16B-aligned when >=0)
            f32x4 b0 = {0.0f, 0.0f, 0.0f, 0.0f};
            f32x4 b1 = {0.0f, 0.0f, 0.0f, 0.0f};
            if (rowok && chunk >= 1 && chunk <= 40) {        // chunk-uniform predicate
                b0 = *(const f32x4*)(g2 + (2 * cp)     * HW + x);
                b1 = *(const f32x4*)(g2 + (2 * cp + 1) * HW + x);
            }
#pragma unroll
            for (int e = 0; e < 4; ++e) {
                union { __bf16 h[2]; unsigned u; } pk;
                pk.h[0] = (__bf16)b0[e];
                pk.h[1] = (__bf16)b1[e];
                *(unsigned*)&Bbuf[(chunk * 4 + e) * LSTR + 2 * cp] = pk.u;
            }
        }
        __syncthreads();

        // ---------------- banded matmul: two 16x16 N-tiles cover the +-4 band ----
        // C0 columns: x in [x0-4, x0+12)  -> pos = x0 + n
        // C1 columns: x in [x0+8, x0+24)  -> pos = x0 + 12 + n
        // B 32x16 as loaded here (same k order as A): lane n (and n+16) holds column n; one contiguous
        // 16-element K run at koff = (lane>=16)*16.
        v8f c0 = {}, c1 = {};
        const __bf16* b0col = Bbuf + (size_t)(x0 + n)      * LSTR;
        const __bf16* b1col = Bbuf + (size_t)(x0 + 12 + n) * LSTR;
        const int     koffb = ((lane >> 4) & 1) * 16;
#pragma unroll
        for (int kc = 0; kc < 4; ++kc) {
            union { v16bf v; bf16x8 h[2]; } ub0, ub1;
            ub0.h[0] = *(const bf16x8*)(b0col + kc * 32 + koffb);
            ub0.h[1] = *(const bf16x8*)(b0col + kc * 32 + koffb + 8);
            ub1.h[0] = *(const bf16x8*)(b1col + kc * 32 + koffb);
            ub1.h[1] = *(const bf16x8*)(b1col + kc * 32 + koffb + 8);
            c0 = __builtin_amdgcn_wmma_f32_16x16x32_bf16(false, afrag[kc],
                                                         false, ub0.v,
                                                         (short)0, c0, false, false);
            c1 = __builtin_amdgcn_wmma_f32_16x16x32_bf16(false, afrag[kc],
                                                         false, ub1.v,
                                                         (short)0, c1, false, false);
        }

        // ---------------- dump C tiles to LDS (branch-free b128 stores) ----------
        // C frag layout: VGPR r, lane L holds element (m = r + (L>=16)*8, n = L&15).
        // Lane L owns a contiguous run of 8 rows of one column -> 2x ds_store_b128.
        // Ctile index: [local col t][row m], t = x - (x0 - 4).  C0 -> t = n,
        // C1 -> t = n + 12 (overlap t=12..15 written twice, identical values).
        {
            float* ct = Ctile + wave * CT_WAVE;
            union { v8f v; f32x4 q[2]; } u0, u1;
            u0.v = c0; u1.v = c1;
            float* p0 = ct + n * CT_STR + madd;           // 16B aligned
            float* p1 = ct + (n + 12) * CT_STR + madd;
            *(f32x4*)(p0)     = u0.q[0];
            *(f32x4*)(p0 + 4) = u0.q[1];
            *(f32x4*)(p1)     = u1.q[0];
            *(f32x4*)(p1 + 4) = u1.q[1];
        }
        __syncthreads();

        // ---------------- coalesced output: fixed x per thread -------------------
        // out[b, dy*9+s4, y, xo] = Ctile[wave(xo)][(xo&15) + s4][xo&15] * scale
        {
            float* outx = out + ((size_t)(b * 81 + dy * 9) * H_DIM + y) * W_DIM + xo;
#pragma unroll
            for (int k = 0; k < 5; ++k) {
                int s4 = sp + 2 * k;      // parity-split planes, all 320 threads busy
                if (s4 < ND) {
                    outx[(size_t)s4 * HW] = ctp[s4 * CT_STR] * scale;
                }
            }
        }
    }
}

extern "C" void kernel_launch(void* const* d_in, const int* in_sizes, int n_in,
                              void* d_out, int out_size, void* d_ws, size_t ws_size,
                              hipStream_t stream) {
    (void)in_sizes; (void)n_in; (void)d_ws; (void)ws_size; (void)out_size;
    const float* in1 = (const float*)d_in[0];
    const float* in2 = (const float*)d_in[1];
    float* out = (float*)d_out;

    dim3 grid(H_DIM, 16);      // (y, b) -> 2048 workgroups
    dim3 block(NTHREADS);      // 10 waves, one 16-pixel tile each
    corr_wmma_kernel<<<grid, block, 0, stream>>>(in1, in2, out);
}
